// Shift_Window_MSA_19430432047558
// MI455X (gfx1250) — matrix operand order repaired and checked by reading, not compiled, not run on hardware
//
#include <hip/hip_runtime.h>
#include <hip/hip_bf16.h>

typedef _Float16 v8h  __attribute__((ext_vector_type(8)));
typedef _Float16 v16h __attribute__((ext_vector_type(16)));
typedef float    v8f  __attribute__((ext_vector_type(8)));
typedef unsigned int u32x4 __attribute__((ext_vector_type(4)));
typedef int          i32x4 __attribute__((ext_vector_type(4)));
typedef int          i32x8 __attribute__((ext_vector_type(8)));

#if defined(__has_include)
#if __has_include(<hip/amd_detail/amd_gfx1250_TDM.h>)
#define SWIN_TDM_6ARG 1
#endif
#endif

#define WIN    7
#define NTOK   49
#define NPAD   64
#define HEADS  6
#define HD     32
#define CCH    192
#define SHIFT3 3
#define HW     56
#define XS     200   // f16 row stride for Xs/Qs/Ks/Os (mult of 8 -> 16B aligned chunks)
#define VT     72    // f16 row stride for Vt and per-wave attn tiles
#define SSTR   68    // f32 row stride for per-wave score tiles
// dynamic LDS starts at offset 0 (no static LDS in kernel); staging aliases Sw/Aw
#define STAGE_OFF ((3 * NPAD * XS + CCH * VT) * 2)   // = 104448 bytes

__device__ __forceinline__ v16h fuse16(v8h lo, v8h hi) {
  return __builtin_shufflevector(lo, hi, 0,1,2,3,4,5,6,7,8,9,10,11,12,13,14,15);
}

__device__ __forceinline__ v8f splat8(float f) {
  v8f r;
#pragma unroll
  for (int i = 0; i < 8; ++i) r[i] = f;
  return r;
}

// A fragment (16x32 f16, MxK) from row-major [m][k] storage (contiguous k = 16*sub + s).
__device__ __forceinline__ v16h load_fragA(const _Float16* base, int m0, int k0,
                                           int stride, int lane) {
  const int sub = lane >> 4;
  const _Float16* p = base + (m0 + (lane & 15)) * stride + k0 + 16 * sub;
  v8h lo = *(const v8h*)p;
  v8h hi = *(const v8h*)(p + 8);
  return fuse16(lo, hi);
}

// B fragment (32x16 f16, KxN) from B^T row-major [n][k] storage.
__device__ __forceinline__ v16h load_fragB(const _Float16* base, int n0, int k0,
                                           int stride, int lane) {
  const int sub = lane >> 4;
  const _Float16* p = base + (n0 + (lane & 15)) * stride + k0 + 16 * sub;
  v8h lo = *(const v8h*)p;
  v8h hi = *(const v8h*)(p + 8);
  return fuse16(lo, hi);
}

__device__ __forceinline__ v8f wmma_f16(v16h a, v16h b, v8f c) {
  return __builtin_amdgcn_wmma_f32_16x16x32_f16(false, a, false, b, (short)0, c,
                                                false, false);
}

__device__ __forceinline__ int swin_reg(int p) { return p < 49 ? 0 : (p < 53 ? 1 : 2); }

// Issue one TDM 2D tile load: global (tile_w elems wide, tile_h rows, f32,
// row stride row_stride elems) -> contiguous LDS at byte offset lds_off.
// D# bitfield packing per CDNA5 ISA ch.8 (group0 128b, group1 256b).
__device__ __forceinline__ void tdm_issue(unsigned lds_off,
                                          unsigned long long gaddr,
                                          unsigned tile_w, unsigned tile_h,
                                          unsigned row_stride) {
  u32x4 g0;
  g0[0] = 1u;                                             // count=1, user mode
  g0[1] = lds_off;                                        // lds_addr (bytes)
  g0[2] = (unsigned)(gaddr & 0xFFFFFFFFull);              // global_addr[31:0]
  g0[3] = (unsigned)((gaddr >> 32) & 0x1FFFFFFull) | 0x80000000u; // addr[56:32] | type=2
  i32x8 g1;
  g1[0] = (int)(2u << 16);                                // data_size = 4B
  g1[1] = (int)((tile_w & 0xFFFFu) << 16);                // tensor_dim0[15:0]
  g1[2] = (int)((tile_w >> 16) & 0xFFFFu) |               // tensor_dim0[31:16]
          (int)((tile_h & 0xFFFFu) << 16);                // tensor_dim1[15:0]
  g1[3] = (int)((tile_h >> 16) & 0xFFFFu) |               // tensor_dim1[31:16]
          (int)((tile_w & 0xFFFFu) << 16);                // tile_dim0
  g1[4] = (int)(tile_h & 0xFFFFu);                        // tile_dim1 (tile_dim2=0)
  g1[5] = (int)row_stride;                                // tensor_dim0_stride[31:0]
  g1[6] = 0;                                              // stride hi / dim1_stride lo
  g1[7] = 0;
  i32x4 z4 = {0, 0, 0, 0};
#ifdef SWIN_TDM_6ARG
  i32x8 z8 = {0, 0, 0, 0, 0, 0, 0, 0};
  __builtin_amdgcn_tensor_load_to_lds(g0, g1, z4, z4, z8, 0);
#else
  __builtin_amdgcn_tensor_load_to_lds(g0, g1, z4, z4, 0);
#endif
}

// Pack an f32 weight matrix [6*32 x ncols] into f16 WMMA B-fragment order.
__global__ void swin_prep_w(const float* __restrict__ w, _Float16* __restrict__ dst,
                            int NT, int ncols) {
  int id = blockIdx.x * 256 + threadIdx.x;
  int total = 6 * NT * 512;
  if (id >= total) return;
  int j    = id & 15;
  int lane = (id >> 4) & 31;
  int nt   = (id >> 9) % NT;
  int kt   = (id >> 9) / NT;
  int k = kt * 32 + 16 * (lane >> 4) + j;
  int n = nt * 16 + (lane & 15);
  dst[id] = (_Float16)w[k * ncols + n];
}

__global__ __launch_bounds__(256) void swin_msa_fused(
    const float* __restrict__ x, const float* __restrict__ qkv_b,
    const float* __restrict__ rel, const float* __restrict__ proj_b,
    const _Float16* __restrict__ wQKV, const _Float16* __restrict__ wP,
    float* __restrict__ out) {
  extern __shared__ _Float16 smem[];
  _Float16* Xs = smem;                        // 64*200 f16 (reused as Os)
  _Float16* Qs = Xs + NPAD * XS;              // 64*200 f16 (q pre-scaled)
  _Float16* Ks = Qs + NPAD * XS;              // 64*200 f16
  _Float16* Vt = Ks + NPAD * XS;              // 192*72 f16, transposed [chan][token]
  float*    Sw = (float*)(Vt + CCH * VT);     // 8 waves * 16*68 f32 scores
  _Float16* Aw = (_Float16*)(Sw + 8 * 16 * SSTR); // 8 waves * 16*72 f16 attn
  _Float16* Os = Xs;
  float*    Xstage = Sw;                      // 49*192 f32 TDM staging (aliases Sw/Aw)

  const int tid  = threadIdx.x;
  const int wave = tid >> 5, lane = tid & 31;
  const int sub  = lane >> 4, ln = lane & 15;
  const int blk  = blockIdx.x;
  const int b    = blk >> 6;
  const int wIdx = blk & 63;
  const int wi   = wIdx >> 3, wj = wIdx & 7;

  // ---- Stage A0: TDM gather of the cyclic-shifted window into f32 staging.
  // Window rows/cols wrap at most once -> <= 2x2 rectangular runs; full-width
  // runs go as one 2D descriptor, narrow runs as one descriptor per row.
  if (wave == 0) {
    int rsplit = 56 - (wi * WIN + SHIFT3);  // first r that wraps (>=7: none)
    int csplit = 56 - (wj * WIN + SHIFT3);
    for (int rr = 0; rr < 2; ++rr) {
      int rbeg = (rr == 0) ? 0 : rsplit;
      int rend = (rr == 0) ? (rsplit < 7 ? rsplit : 7) : 7;
      if (rbeg >= rend) continue;
      int gr0 = wi * WIN + rbeg + SHIFT3; if (gr0 >= HW) gr0 -= HW;
      for (int cc = 0; cc < 2; ++cc) {
        int cbeg = (cc == 0) ? 0 : csplit;
        int cend = (cc == 0) ? (csplit < 7 ? csplit : 7) : 7;
        if (cbeg >= cend) continue;
        int gc0 = wj * WIN + cbeg + SHIFT3; if (gc0 >= HW) gc0 -= HW;
        int rcount = rend - rbeg, ccount = cend - cbeg;
        unsigned long long gbase =
            (unsigned long long)(const void*)x +
            (unsigned long long)(((b * HW + gr0) * HW + gc0) * CCH) * 4ull;
        if (ccount == WIN) {
          tdm_issue(STAGE_OFF + (unsigned)((rbeg * WIN + cbeg) * CCH * 4),
                    gbase, (unsigned)(ccount * CCH), (unsigned)rcount,
                    (unsigned)(HW * CCH));
        } else {
          for (int r = 0; r < rcount; ++r) {
            tdm_issue(STAGE_OFF + (unsigned)(((rbeg + r) * WIN + cbeg) * CCH * 4),
                      gbase + (unsigned long long)(r * HW * CCH) * 4ull,
                      (unsigned)(ccount * CCH), 1u, (unsigned)(HW * CCH));
          }
        }
      }
    }
    __builtin_amdgcn_s_wait_tensorcnt((short)0);
  }
  __syncthreads();

  // ---- Stage A1: staging f32 -> f16 Xs (zero-pad rows 49..63)
  for (int t = tid; t < NPAD * CCH; t += 256) {
    int row = t / CCH, ch = t - row * CCH;
    Xs[row * XS + ch] =
        (row < NTOK) ? (_Float16)Xstage[row * CCH + ch] : (_Float16)0.f;
  }
  __syncthreads();

  // ---- Stage B: QKV GEMM  (64x192) x (192x576), 144 tiles over 8 waves
  for (int t = wave; t < 144; t += 8) {
    int nt = t >> 2, mt = t & 3;
    v8f acc = splat8(qkv_b[nt * 16 + ln]);
#pragma unroll
    for (int kt = 0; kt < 6; ++kt) {
      v16h a  = load_fragA(Xs, mt * 16, kt * 32, XS, lane);
      v16h bf = *(const v16h*)(wQKV + (((kt * 36 + nt) << 5) + lane) * 16);
      if (kt < 5)
        __builtin_prefetch(wQKV + ((((kt + 1) * 36 + nt) << 5) + lane) * 16, 0, 3);
      acc = wmma_f16(a, bf, acc);
    }
    int col = nt * 16 + ln;
    int rb  = mt * 16 + 8 * sub;
    if (col < CCH) {
#pragma unroll
      for (int i = 0; i < 8; ++i)
        Qs[(rb + i) * XS + col] = (_Float16)(acc[i] * 0.17677669529663687f);
    } else if (col < 2 * CCH) {
#pragma unroll
      for (int i = 0; i < 8; ++i)
        Ks[(rb + i) * XS + (col - CCH)] = (_Float16)acc[i];
    } else {
#pragma unroll
      for (int i = 0; i < 8; ++i)
        Vt[(col - 2 * CCH) * VT + rb + i] = (_Float16)acc[i];
    }
  }
  __syncthreads();

  // ---- Stage C: attention. 24 (head,row-tile) tasks, exactly 3 per wave.
  float*    Swv = Sw + wave * 16 * SSTR;
  _Float16* Awv = Aw + wave * 16 * VT;
  for (int t = wave; t < 24; t += 8) {
    int h = t >> 2, mt = t & 3;
    v8f s[4];
#pragma unroll
    for (int ct = 0; ct < 4; ++ct) s[ct] = splat8(0.f);
    v16h aq = load_fragA(Qs, mt * 16, h * HD, XS, lane);
#pragma unroll
    for (int ct = 0; ct < 4; ++ct) {
      v16h bk = load_fragB(Ks, ct * 16, h * HD, XS, lane);
      s[ct] = wmma_f16(aq, bk, s[ct]);
    }
#pragma unroll
    for (int ct = 0; ct < 4; ++ct)
#pragma unroll
      for (int i = 0; i < 8; ++i)
        Swv[(8 * sub + i) * SSTR + ct * 16 + ln] = s[ct][i];
    asm volatile("s_wait_dscnt 0" ::: "memory");

    // lane-pair softmax: 2 lanes per row, 32 columns each
    int rloc = lane >> 1, half = lane & 1;
    int tokm = mt * 16 + rloc;
    float vals[32];
    float mx = -3.0e38f;
#pragma unroll
    for (int j = 0; j < 32; ++j) {
      int tokn = half * 32 + j;
      float v = (tokn < NTOK) ? Swv[rloc * SSTR + tokn] : -3.0e38f;
      vals[j] = v;
      mx = fmaxf(mx, v);
    }
    mx = fmaxf(mx, __shfl_xor(mx, 1, 32));
    float sum = 0.f;
#pragma unroll
    for (int j = 0; j < 32; ++j) {
      int tokn = half * 32 + j;
      if (tokn < NTOK) { vals[j] = __expf(vals[j] - mx); sum += vals[j]; }
      else             { vals[j] = 0.f; }
    }
    sum += __shfl_xor(sum, 1, 32);
    float inv = 1.0f / sum;

    // post-softmax relative bias + shift mask (reference adds them AFTER softmax)
    int r1 = tokm / WIN, c1 = tokm - r1 * WIN;
    int labM = swin_reg(wi * WIN + r1) * 3 + swin_reg(wj * WIN + c1);
#pragma unroll
    for (int j = 0; j < 32; ++j) {
      int tokn = half * 32 + j;
      float o = 0.f;
      if (tokm < NTOK && tokn < NTOK) {
        int r2 = tokn / WIN, c2 = tokn - r2 * WIN;
        int idx = (r1 - r2 + WIN - 1) * (2 * WIN - 1) + (c1 - c2 + WIN - 1);
        int labN = swin_reg(wi * WIN + r2) * 3 + swin_reg(wj * WIN + c2);
        o = vals[j] * inv + rel[idx * HEADS + h] + ((labM == labN) ? 0.f : -100.f);
      }
      Awv[rloc * VT + tokn] = (_Float16)o;
    }
    asm volatile("s_wait_dscnt 0" ::: "memory");

    // O = attn (16x64) x V (64x32); V transposed in LDS -> contiguous B frags
    v8f o0 = splat8(0.f), o1 = splat8(0.f);
#pragma unroll
    for (int k2 = 0; k2 < 2; ++k2) {
      v16h aa = load_fragA(Awv, 0, k2 * 32, VT, lane);
      v16h b0 = load_fragB(Vt, h * HD,      k2 * 32, VT, lane);
      v16h b1 = load_fragB(Vt, h * HD + 16, k2 * 32, VT, lane);
      o0 = wmma_f16(aa, b0, o0);
      o1 = wmma_f16(aa, b1, o1);
    }
    int rb = mt * 16 + 8 * sub;
#pragma unroll
    for (int i = 0; i < 8; ++i) {
      Os[(rb + i) * XS + h * HD + ln]      = (_Float16)o0[i];
      Os[(rb + i) * XS + h * HD + 16 + ln] = (_Float16)o1[i];
    }
  }
  __syncthreads();

  // ---- Stage D: proj GEMM (64x192)x(192x192) + window-reverse + un-shift scatter
  for (int t = wave; t < 48; t += 8) {
    int nt = t >> 2, mt = t & 3;
    v8f acc = splat8(proj_b[nt * 16 + ln]);
#pragma unroll
    for (int kt = 0; kt < 6; ++kt) {
      v16h a  = load_fragA(Os, mt * 16, kt * 32, XS, lane);
      v16h bw = *(const v16h*)(wP + (((kt * 12 + nt) << 5) + lane) * 16);
      acc = wmma_f16(a, bw, acc);
    }
    int rb  = mt * 16 + 8 * sub;
    int col = nt * 16 + ln;
#pragma unroll
    for (int i = 0; i < 8; ++i) {
      int tok = rb + i;
      if (tok < NTOK) {
        int r = tok / WIN, c = tok - r * WIN;
        int gr = wi * WIN + r + SHIFT3; if (gr >= HW) gr -= HW;
        int gc = wj * WIN + c + SHIFT3; if (gc >= HW) gc -= HW;
        out[((b * HW + gr) * HW + gc) * CCH + col] = acc[i];
      }
    }
  }
}

extern "C" void kernel_launch(void* const* d_in, const int* in_sizes, int n_in,
                              void* d_out, int out_size, void* d_ws, size_t ws_size,
                              hipStream_t stream) {
  const float* x      = (const float*)d_in[0];
  const float* qkv_w  = (const float*)d_in[1];
  const float* qkv_b  = (const float*)d_in[2];
  const float* rel    = (const float*)d_in[3];
  const float* proj_w = (const float*)d_in[4];
  const float* proj_b = (const float*)d_in[5];
  float* out = (float*)d_out;

  _Float16* wQKV = (_Float16*)d_ws;          // 192*576 f16 = 221184 B
  _Float16* wP   = wQKV + 192 * 576;         // 192*192 f16 =  73728 B

  swin_prep_w<<<432, 256, 0, stream>>>(qkv_w, wQKV, 36, 576);
  swin_prep_w<<<144, 256, 0, stream>>>(proj_w, wP, 12, 192);

  const size_t shmem =
      (size_t)(3 * NPAD * XS + CCH * VT) * 2 +   // Xs/Qs/Ks + Vt (f16)
      (size_t)(8 * 16 * SSTR) * 4 +              // score tiles (f32)
      (size_t)(8 * 16 * VT) * 2;                 // attn tiles (f16)  = 157696 B

  swin_msa_fused<<<dim3(4096), dim3(256), shmem, stream>>>(
      x, qkv_b, rel, proj_b, wQKV, wP, out);
}
